// PCTv2_SegReg_54889682043700
// MI455X (gfx1250) — matrix operand order repaired and checked by reading, not compiled, not run on hardware
//
#include <hip/hip_runtime.h>
#include <hip/hip_bf16.h>
#include <cstdint>

typedef __attribute__((ext_vector_type(16))) _Float16 v16h;
typedef __attribute__((ext_vector_type(8)))  _Float16 v8h;
typedef __attribute__((ext_vector_type(8)))  float    v8f;
typedef unsigned int u32x4 __attribute__((ext_vector_type(4)));
typedef int          i32x4 __attribute__((ext_vector_type(4)));
typedef int          i32x8 __attribute__((ext_vector_type(8)));

#define BB 32
#define NN 16384
#define CC 48
#define K_BN_EPS 1e-5f
#define TILES_PER_BLOCK 8

// ---- workspace layout (float offsets) ----
#define WS_XMAX   0        // 32*48 = 1536
#define WS_XMEAN  1536     // 1536
#define WS_BIAS1  3072     // 2*32*96 = 6144
#define WS_BIAS2  9216     // 96 (pad 128)
#define WS_BIAS3  9344     // 32 (pad 64)
#define WS_ACC    9408     // 32*16*4 = 2048
#define WS_HALF   11456    // start of f16 region (byte 45824, 32B aligned)
// ---- f16 region offsets (half offsets) ----
#define WH_W1 0            // 2*96*64 = 12288
#define WH_W2 12288        // 2*48*96 = 9216
#define WH_W3 21504        // 2*16*64 = 2048

struct HeadP {
  const float *w1,*b1,*g1,*be1,*m1,*v1;
  const float *w2,*b2,*g2,*be2,*m2,*v2;
  const float *w3,*b3;
};

// ---------------- kernel 1: per-(b,c) max / mean over N ----------------
__global__ void k_reduce(const float* __restrict__ feat, float* __restrict__ ws) {
  __shared__ float smax[256];
  __shared__ float ssum[256];
  const int bc = blockIdx.x;                 // 0 .. B*C-1
  const float* p = feat + (size_t)bc * NN;
  float mx = -3.4e38f, sm = 0.f;
  for (int i = threadIdx.x; i < NN; i += 256) {
    float v = p[i];
    mx = fmaxf(mx, v);
    sm += v;
  }
  smax[threadIdx.x] = mx; ssum[threadIdx.x] = sm;
  __syncthreads();
  for (int s = 128; s > 0; s >>= 1) {
    if (threadIdx.x < s) {
      smax[threadIdx.x] = fmaxf(smax[threadIdx.x], smax[threadIdx.x + s]);
      ssum[threadIdx.x] += ssum[threadIdx.x + s];
    }
    __syncthreads();
  }
  if (threadIdx.x == 0) {
    ws[WS_XMAX + bc]  = smax[0];
    ws[WS_XMEAN + bc] = ssum[0] * (1.0f / NN);
  }
}

// ---------------- kernel 2: fold BN into f16 weights, zero accumulators ----------------
__global__ void k_fold(HeadP hp0, HeadP hp1, float* __restrict__ ws) {
  _Float16* hb = (_Float16*)(ws + WS_HALF);
  const int t = threadIdx.x;

  for (int idx = t; idx < 2*96*64; idx += 256) {          // W1eff [head][96][64]
    int head = idx / (96*64); int rem = idx % (96*64);
    int o = rem / 64, k = rem % 64;
    HeadP hp = head ? hp1 : hp0;
    float s = hp.g1[o] * rsqrtf(hp.v1[o] + K_BN_EPS);
    float v = (k < 48) ? hp.w1[o*144 + k] * s : 0.f;
    hb[WH_W1 + idx] = (_Float16)v;
  }
  for (int idx = t; idx < 2*48*96; idx += 256) {          // W2eff [head][48][96]
    int head = idx / (48*96); int rem = idx % (48*96);
    int o = rem / 96, k = rem % 96;
    HeadP hp = head ? hp1 : hp0;
    float s = hp.g2[o] * rsqrtf(hp.v2[o] + K_BN_EPS);
    hb[WH_W2 + idx] = (_Float16)(hp.w2[o*96 + k] * s);
  }
  for (int idx = t; idx < 2*48; idx += 256) {             // bias2 [head][48]
    int head = idx / 48, o = idx % 48;
    HeadP hp = head ? hp1 : hp0;
    float s = hp.g2[o] * rsqrtf(hp.v2[o] + K_BN_EPS);
    ws[WS_BIAS2 + idx] = (hp.b2[o] - hp.m2[o]) * s + hp.be2[o];
  }
  for (int idx = t; idx < 2*16*64; idx += 256) {          // W3eff [head][16][64]
    int head = idx / (16*64); int rem = idx % (16*64);
    int o = rem / 64, k = rem % 64;
    HeadP hp = head ? hp1 : hp0;
    int rows = head ? 16 : 3;
    float v = (o < rows && k < 48) ? hp.w3[o*48 + k] : 0.f;
    hb[WH_W3 + idx] = (_Float16)v;
  }
  for (int idx = t; idx < 2*16; idx += 256) {             // bias3 [head][16]
    int head = idx / 16, o = idx % 16;
    HeadP hp = head ? hp1 : hp0;
    int rows = head ? 16 : 3;
    ws[WS_BIAS3 + idx] = (o < rows) ? hp.b3[o] : 0.f;
  }
  for (int idx = t; idx < 2048; idx += 256) ws[WS_ACC + idx] = 0.f;
}

// ---------------- kernel 3: per-batch effective bias1 ----------------
__global__ void k_bias1(HeadP hp0, HeadP hp1, float* __restrict__ ws) {
  const int head = blockIdx.x >> 5;
  const int b    = blockIdx.x & 31;
  const int o    = threadIdx.x;          // 96 threads
  HeadP hp = head ? hp1 : hp0;
  float s   = hp.g1[o] * rsqrtf(hp.v1[o] + K_BN_EPS);
  float acc = (hp.b1[o] - hp.m1[o]) * s + hp.be1[o];
  const float* xm = ws + WS_XMAX  + b*48;
  const float* xe = ws + WS_XMEAN + b*48;
  const float* w  = hp.w1 + o*144;
  for (int c = 0; c < 48; ++c)
    acc += s * (w[48 + c] * xm[c] + w[96 + c] * xe[c]);
  ws[WS_BIAS1 + (head*32 + b)*96 + o] = acc;
}

// A fragment: 16x32 f16 from row-major W (leading dim ld).
__device__ __forceinline__ v16h load_afrag(const _Float16* __restrict__ W, int ld,
                                           int m, int koff, int h) {
  const v8h a0 = *reinterpret_cast<const v8h*>(W + m*ld + koff + 16*h);
  const v8h a1 = *reinterpret_cast<const v8h*>(W + m*ld + koff + 16*h + 8);
  v16h r;
#pragma unroll
  for (int i = 0; i < 8; ++i) { r[i] = a0[i]; r[i+8] = a1[i]; }
  return r;
}

__device__ __forceinline__ v8f wmma_f16(v16h a, v16h b, v8f c) {
  return __builtin_amdgcn_wmma_f32_16x16x32_f16(false, a, false, b, (short)0, c, false, false);
}

// ---- Tensor Data Mover: async 2-D tile (48 rows x 64 cols f32, row stride NN)
//      from global into LDS at lds_byte_off.  D# per CDNA5 ISA.
__device__ __forceinline__ void tdm_load_tile(const float* gsrc, uint32_t lds_byte_off) {
  const uint64_t ga = (uint64_t)(uintptr_t)gsrc;
  u32x4 g0;
  g0[0] = 1u;                                             // count=1, user descriptor
  g0[1] = lds_byte_off;                                   // lds_addr
  g0[2] = (uint32_t)ga;                                   // global_addr[31:0]
  g0[3] = (uint32_t)((ga >> 32) & 0x01FFFFFFu) | (2u << 30); // global_addr[56:32] | type=2
  i32x8 g1;
  g1[0] = 0x00020000;                 // workgroup_mask=0, data_size=2 (4 bytes)
  g1[1] = (int)(16384u << 16);        // tensor_dim0[15:0]=16384 in bits 63:48
  g1[2] = (int)(48u << 16);           // tensor_dim0[31:16]=0 | tensor_dim1[15:0]=48
  g1[3] = (int)(64u << 16);           // tensor_dim1[31:16]=0 | tile_dim0=64
  g1[4] = 48;                         // tile_dim1=48, tile_dim2=0
  g1[5] = 16384;                      // tensor_dim0_stride[31:0]
  g1[6] = 0;                          // stride hi | tensor_dim1_stride lo
  g1[7] = 0;
  i32x4 z4 = {0, 0, 0, 0};
#if __clang_major__ >= 23
  i32x8 z8 = {0, 0, 0, 0, 0, 0, 0, 0};
  __builtin_amdgcn_tensor_load_to_lds(g0, g1, z4, z4, z8, 0);
#else
  __builtin_amdgcn_tensor_load_to_lds(g0, g1, z4, z4, 0);
#endif
}

// ---------------- kernel 4: main fused MLP + pooling (TDM double-buffered) ----------------
__global__ __launch_bounds__(128) void k_main(const float* __restrict__ feat,
                                              const float* __restrict__ xyz,
                                              float* __restrict__ ws,
                                              float* __restrict__ out_xr,
                                              float* __restrict__ out_xs) {
  __shared__ alignas(16) float    sh_stage[2][48*64];   // TDM landing buffers (f32 row-major)
  __shared__ alignas(32) _Float16 sh_x [64][64];        // f16 [col][K], K padded to 64
  __shared__ alignas(32) _Float16 sh_h [4][16][96];
  __shared__ alignas(32) _Float16 sh_h2[4][16][64];
  __shared__ float sh_cls[4][16][16];
  __shared__ float sh_reg[4][16][4];
  __shared__ float sh_acc[64];

  const int tid  = threadIdx.x;
  const int wave = tid >> 5, lane = tid & 31;
  const int lm   = lane & 15, lh = lane >> 4;
  const int b        = blockIdx.x >> 5;                 // 32 blocks per batch
  const int tileBase = (blockIdx.x & 31) * TILES_PER_BLOCK;

  const _Float16* hb = (const _Float16*)(ws + WS_HALF);
  const _Float16* W1 = hb + WH_W1;
  const _Float16* W2 = hb + WH_W2;
  const _Float16* W3 = hb + WH_W3;
  const float* fbat = feat + (size_t)b * CC * NN;

#pragma unroll
  for (int i = 0; i < 8; ++i) sh_h2[wave][lm][48 + 8*lh + i] = (_Float16)0.f;
  for (int idx = tid; idx < 16*64; idx += 128) {        // zero K-pad of sh_x once
    int k = 48 + (idx >> 6), j = idx & 63;
    sh_x[j][k] = (_Float16)0.f;
  }
  if (tid < 64) sh_acc[tid] = 0.f;

  // prime the pipeline: async-load tile 0
  if (wave == 0)
    tdm_load_tile(fbat + tileBase*64, (uint32_t)(uintptr_t)(void*)&sh_stage[0][0]);
  __syncthreads();

  for (int t = 0; t < TILES_PER_BLOCK; ++t) {
    const int buf  = t & 1;
    const int n0   = (tileBase + t) * 64;
    const int ncol = n0 + wave*16 + lm;

    if (wave == 0) {
      __builtin_amdgcn_s_wait_tensorcnt(0);             // tile t landed in sh_stage[buf]
    }
    __syncthreads();
    if (t + 1 < TILES_PER_BLOCK && wave == 0)           // overlap next DMA with compute
      tdm_load_tile(fbat + (tileBase + t + 1)*64,
                    (uint32_t)(uintptr_t)(void*)&sh_stage[buf ^ 1][0]);

    // convert landed f32 row-major tile -> f16 column-K staging
    for (int idx = tid; idx < 48*64; idx += 128) {
      int c = idx >> 6, j = idx & 63;
      sh_x[j][c] = (_Float16)sh_stage[buf][c*64 + j];
    }
    __syncthreads();

    for (int head = 0; head < 2; ++head) {
      const _Float16* W1h = W1 + head*96*64;
      const _Float16* W2h = W2 + head*48*96;
      const _Float16* W3h = W3 + head*16*64;
      const float* bias1 = ws + WS_BIAS1 + (head*32 + b)*96;
      const float* bias2 = ws + WS_BIAS2 + head*48;
      const float* bias3 = ws + WS_BIAS3 + head*16;

      // ---- layer 1: (96 x 64) @ (64 x 16) ----
#pragma unroll
      for (int mt = 0; mt < 6; ++mt) {
        v8f c;
#pragma unroll
        for (int r = 0; r < 8; ++r) c[r] = bias1[mt*16 + 8*lh + r];
#pragma unroll
        for (int ks = 0; ks < 2; ++ks) {
          v16h a  = load_afrag(W1h, 64, mt*16 + lm, ks*32, lh);
          v16h bm = *reinterpret_cast<const v16h*>(&sh_x[wave*16 + lm][ks*32 + 16*lh]);
          c = wmma_f16(a, bm, c);
        }
#pragma unroll
        for (int r = 0; r < 8; ++r)
          sh_h[wave][lm][mt*16 + 8*lh + r] = (_Float16)fmaxf(c[r], 0.f);
      }
      __syncthreads();

      // ---- layer 2: (48 x 96) @ (96 x 16) ----
#pragma unroll
      for (int mt = 0; mt < 3; ++mt) {
        v8f c;
#pragma unroll
        for (int r = 0; r < 8; ++r) c[r] = bias2[mt*16 + 8*lh + r];
#pragma unroll
        for (int ks = 0; ks < 3; ++ks) {
          v16h a  = load_afrag(W2h, 96, mt*16 + lm, ks*32, lh);
          v16h bm = *reinterpret_cast<const v16h*>(&sh_h[wave][lm][ks*32 + 16*lh]);
          c = wmma_f16(a, bm, c);
        }
#pragma unroll
        for (int r = 0; r < 8; ++r)
          sh_h2[wave][lm][mt*16 + 8*lh + r] = (_Float16)fmaxf(c[r], 0.f);
      }
      __syncthreads();

      // ---- layer 3: (16 x 64) @ (64 x 16) ----
      {
        v8f c;
#pragma unroll
        for (int r = 0; r < 8; ++r) c[r] = bias3[8*lh + r];
#pragma unroll
        for (int ks = 0; ks < 2; ++ks) {
          v16h a  = load_afrag(W3h, 64, lm, ks*32, lh);
          v16h bm = *reinterpret_cast<const v16h*>(&sh_h2[wave][lm][ks*32 + 16*lh]);
          c = wmma_f16(a, bm, c);
        }
        if (head == 0) {
          if (lh == 0) {
#pragma unroll
            for (int r = 0; r < 3; ++r) {
              float v = c[r];
              out_xr[(size_t)b*3*NN + (size_t)r*NN + ncol] = v;
              sh_reg[wave][lm][r] = v;
            }
          }
        } else {
#pragma unroll
          for (int r = 0; r < 8; ++r) {
            int row = 8*lh + r;
            out_xs[(size_t)b*16*NN + (size_t)row*NN + ncol] = c[r];
            sh_cls[wave][lm][row] = c[r];
          }
        }
      }
      __syncthreads();
    }

    // ---- softmax, mask, weighted pose accumulation (one point per lane, lh==0) ----
    if (lh == 0) {
      float cls[16];
      float mxv = -3.4e38f;
#pragma unroll
      for (int k = 0; k < 16; ++k) { cls[k] = sh_cls[wave][lm][k]; mxv = fmaxf(mxv, cls[k]); }
      float sum = 0.f;
#pragma unroll
      for (int k = 0; k < 16; ++k) { cls[k] = __expf(cls[k] - mxv); sum += cls[k]; }
      float inv = 1.f / sum;
      float pmax = 0.f;
#pragma unroll
      for (int k = 0; k < 16; ++k) { cls[k] *= inv; pmax = fmaxf(pmax, cls[k]); }
      bool mask = (cls[15] < 0.6f) && (pmax > 0.6f);
      if (mask) {
        const float* xp = xyz + (size_t)b*NN*3 + (size_t)ncol*3;
        float p0 = sh_reg[wave][lm][0] + xp[0];
        float p1 = sh_reg[wave][lm][1] + xp[1];
        float p2 = sh_reg[wave][lm][2] + xp[2];
#pragma unroll
        for (int k = 0; k < 16; ++k) {
          float w = cls[k];
          atomicAdd(&sh_acc[k*4 + 0], w);
          atomicAdd(&sh_acc[k*4 + 1], w * p0);
          atomicAdd(&sh_acc[k*4 + 2], w * p1);
          atomicAdd(&sh_acc[k*4 + 3], w * p2);
        }
      }
    }
    __syncthreads();
  }

  if (tid < 64) atomicAdd(&ws[WS_ACC + b*64 + tid], sh_acc[tid]);
}

// ---------------- kernel 5: finalize pose_r and class_sum_norm ----------------
__global__ void k_final(const float* __restrict__ ws, float* __restrict__ out) {
  const int b = blockIdx.x;
  const int t = threadIdx.x;               // 64 threads
  __shared__ float cs[16];
  const float* acc = ws + WS_ACC + b*64;
  if (t < 16) cs[t] = acc[t*4 + 0];
  __syncthreads();
  if (t < 16) {
    float total = 0.f;
#pragma unroll
    for (int k = 0; k < 16; ++k) total += cs[k];
    out[1440 + b*16 + t] = cs[t] / total;
  }
  if (t < 48) {
    int k = t / 3, d = t % 3;
    if (k < 15) {
      float denom = acc[k*4 + 0] + 1e-8f;
      out[b*45 + k*3 + d] = acc[k*4 + 1 + d] / denom;
    }
  }
}

extern "C" void kernel_launch(void* const* d_in, const int* in_sizes, int n_in,
                              void* d_out, int out_size, void* d_ws, size_t ws_size,
                              hipStream_t stream) {
  const float* xyz  = (const float*)d_in[0];
  const float* feat = (const float*)d_in[1];
  HeadP hp0 = { (const float*)d_in[2],  (const float*)d_in[3],  (const float*)d_in[4],
                (const float*)d_in[5],  (const float*)d_in[6],  (const float*)d_in[7],
                (const float*)d_in[8],  (const float*)d_in[9],  (const float*)d_in[10],
                (const float*)d_in[11], (const float*)d_in[12], (const float*)d_in[13],
                (const float*)d_in[14], (const float*)d_in[15] };
  HeadP hp1 = { (const float*)d_in[16], (const float*)d_in[17], (const float*)d_in[18],
                (const float*)d_in[19], (const float*)d_in[20], (const float*)d_in[21],
                (const float*)d_in[22], (const float*)d_in[23], (const float*)d_in[24],
                (const float*)d_in[25], (const float*)d_in[26], (const float*)d_in[27],
                (const float*)d_in[28], (const float*)d_in[29] };
  float* ws  = (float*)d_ws;
  float* out = (float*)d_out;
  float* out_xr = out + 1952;                    // 1440 pose_r + 512 class_sum_norm
  float* out_xs = out + 1952 + 3*BB*NN;          // after x_r

  k_reduce<<<BB*CC, 256, 0, stream>>>(feat, ws);
  k_fold  <<<1,     256, 0, stream>>>(hp0, hp1, ws);
  k_bias1 <<<64,     96, 0, stream>>>(hp0, hp1, ws);
  k_main  <<<BB*32, 128, 0, stream>>>(feat, xyz, ws, out_xr, out_xs);
  k_final <<<BB,     64, 0, stream>>>(ws, out);
}
